// CAM_Module_11115375362187
// MI455X (gfx1250) — matrix operand order repaired and checked by reading, not compiled, not run on hardware
//
#include <hip/hip_runtime.h>
#include <stdint.h>

// ---------------------------------------------------------------------------
// CAM_Module (channel self-attention) for MI455X (gfx1250), wave32 + WMMA.
//   x: (16, 512, 64, 64) fp32; gamma: (1,) fp32
//   E = X X^T per batch; attention = softmax(rowmax - E) == softmax(-E);
//   out = gamma * (attention @ X) + x
// Pipeline:
//   k0: convert x -> bf16 X (row-major) and bf16 X^T (for GEMM2 B operand)
//   k1: E = X X^T          (bf16 WMMA 16x16x32, f32 accum, 128x128 tiles)
//   k2: attention = softmax(-E) row-wise, written as bf16
//   k3: out = gamma*(A @ X) + x  (bf16 WMMA, fused epilogue)
// GEMM staging is a pure byte copy -> double-buffered LDS via gfx1250
// async-to-LDS (global_load_async_to_lds_b128 + s_wait_asynccnt).
// ---------------------------------------------------------------------------

typedef __attribute__((ext_vector_type(16))) __bf16          v16bf;
typedef __attribute__((ext_vector_type(8)))  float           v8f;
typedef __attribute__((ext_vector_type(8)))  unsigned short  us8;
typedef __attribute__((ext_vector_type(4)))  unsigned short  us4;
typedef __attribute__((ext_vector_type(4)))  float           f4;
typedef int v4i32 __attribute__((vector_size(16)));   // matches builtin param

union Frag { v16bf v; us8 h[2]; };

static constexpr int CB  = 16;    // batches
static constexpr int CC  = 512;   // channels
static constexpr int CN  = 4096;  // H*W
static constexpr int BMT = 128;   // block tile M
static constexpr int BNT = 128;   // block tile N
static constexpr int BK  = 32;    // K step (one bf16 WMMA K)
static constexpr int BKP = 40;    // padded LDS row: 80B, 16B-aligned, conflict-free
static constexpr int TSZ = BMT * BKP;   // elems per staged tile
static constexpr int BUF = 2 * TSZ;     // A-tile + B-tile per buffer

#if defined(__has_builtin)
#if __has_builtin(__builtin_amdgcn_global_load_async_to_lds_b128)
#define HAVE_ASYNC_LDS 1
#endif
#endif

__device__ __forceinline__ unsigned short f2bf(float f) {
    union { float f; uint32_t u; } c; c.f = f;
    uint32_t u = c.u;
    u += 0x7FFFu + ((u >> 16) & 1u);   // round-to-nearest-even
    return (unsigned short)(u >> 16);
}

__device__ __forceinline__ us4 cvt4(f4 a) {
    us4 r; r.x = f2bf(a.x); r.y = f2bf(a.y); r.z = f2bf(a.z); r.w = f2bf(a.w);
    return r;
}

// 16-byte global -> LDS copy (async on gfx1250).
__device__ __forceinline__ void copy16(const unsigned short* g, unsigned short* l) {
#ifdef HAVE_ASYNC_LDS
    __builtin_amdgcn_global_load_async_to_lds_b128(
        (__attribute__((address_space(1))) v4i32*)g,
        (__attribute__((address_space(3))) v4i32*)l, 0, 0);
#else
    *(us8*)l = *(const us8*)g;    // global_load_b128 + ds_store_b128
#endif
}

__device__ __forceinline__ void async_wait() {
#ifdef HAVE_ASYNC_LDS
#if __has_builtin(__builtin_amdgcn_s_wait_asynccnt)
    __builtin_amdgcn_s_wait_asynccnt(0);
#else
    asm volatile("s_wait_asynccnt 0x0" ::: "memory");
#endif
#endif
}

// ---------------------------------------------------------------------------
// Kernel 0: x (fp32, B x C x N) -> xbf (bf16, same layout) and xt (bf16, B x N x C)
// 64x64 tiles, LDS transpose. grid (CN/64, CC/64, CB), block 256.
// ---------------------------------------------------------------------------
__global__ __launch_bounds__(256)
void cam_convert_kernel(const float* __restrict__ x,
                        unsigned short* __restrict__ xbf,
                        unsigned short* __restrict__ xt) {
    const int b  = blockIdx.z;
    const int c0 = blockIdx.y * 64;
    const int n0 = blockIdx.x * 64;
    const float*    xb   = x   + (size_t)b * CC * CN;
    unsigned short* xbfb = xbf + (size_t)b * CC * CN;
    unsigned short* xtb  = xt  + (size_t)b * CN * CC;

    __shared__ unsigned short tl[64][72];
    const int t = threadIdx.x;

    #pragma unroll
    for (int i = 0; i < 4; ++i) {
        const int idx = t + 256 * i;     // 0..1023
        const int r   = idx >> 4;        // channel row 0..63
        const int c4  = idx & 15;        // float4 column
        const f4 v = *(const f4*)(xb + (size_t)(c0 + r) * CN + n0 + c4 * 4);
        const us4 h = cvt4(v);
        *(us4*)(xbfb + (size_t)(c0 + r) * CN + n0 + c4 * 4) = h;
        *(us4*)&tl[r][c4 * 4] = h;
    }
    __syncthreads();
    #pragma unroll
    for (int i = 0; i < 4; ++i) {
        const int idx = t + 256 * i;
        const int n   = idx >> 4;        // spatial row 0..63
        const int cc4 = idx & 15;
        us4 h;
        h.x = tl[cc4 * 4 + 0][n];
        h.y = tl[cc4 * 4 + 1][n];
        h.z = tl[cc4 * 4 + 2][n];
        h.w = tl[cc4 * 4 + 3][n];
        *(us4*)(xtb + (size_t)(n0 + n) * CC + c0 + cc4 * 4) = h;
    }
}

// ---------------------------------------------------------------------------
// Shared GEMM core: C(128x128,f32) += A(128xK,bf16) * B^T-style(128xK,bf16)
// Both operands staged as K-contiguous rows (A rows = M, B rows = N).
// 256 threads = 8 waves in 4x2 grid; wave = 32 rows x 64 cols = 8 WMMAs/step.
// Double-buffered LDS; staging is a pure 16B copy (async path).
// ---------------------------------------------------------------------------
template<int KEXT>
__device__ __forceinline__ void mm_core(const unsigned short* __restrict__ gA, int ldA,
                                        const unsigned short* __restrict__ gB, int ldB,
                                        unsigned short* smem, v8f (&acc)[2][4]) {
    const int t    = threadIdx.x;
    const int lane = t & 31;
    const int w    = t >> 5;        // 0..7
    const int wm   = w & 3;         // row strip of 32
    const int wn   = w >> 2;        // col half of 64
    const int half = lane >> 4;
    const int l16  = lane & 15;

    auto stage = [&](int k0, int p) {
        unsigned short* As = smem + p * BUF;
        unsigned short* Bs = As + TSZ;
        #pragma unroll
        for (int i = 0; i < 2; ++i) {
            const int idx = t + 256 * i;   // 0..511 chunks of 8 bf16
            const int r   = idx >> 2;      // row 0..127
            const int c8  = idx & 3;       // 16B chunk in K slab
            copy16(gA + (size_t)r * ldA + k0 + c8 * 8, As + r * BKP + c8 * 8);
            copy16(gB + (size_t)r * ldB + k0 + c8 * 8, Bs + r * BKP + c8 * 8);
        }
    };

    stage(0, 0);
    async_wait();
    __syncthreads();

    int p = 0;
    for (int k0 = 0; k0 < KEXT; k0 += BK) {
        if (k0 + BK < KEXT) stage(k0 + BK, p ^ 1);   // overlap with compute

        const unsigned short* As = smem + p * BUF;
        const unsigned short* Bs = As + TSZ;

        // A fragments (16x32 bf16): lane half -> K {0-7,16-23} / {8-15,24-31}
        Frag a[2];
        #pragma unroll
        for (int m = 0; m < 2; ++m) {
            const unsigned short* ar = As + (wm * 32 + m * 16 + l16) * BKP;
            a[m].h[0] = *(const us8*)(ar + 8 * half);
            a[m].h[1] = *(const us8*)(ar + 16 + 8 * half);
        }
        #pragma unroll
        for (int s = 0; s < 4; ++s) {
            // B fragment (32x16): lane = column n, half -> K {0-7,16-23} / {8-15,24-31}
            Frag bf;
            const unsigned short* br = Bs + (wn * 64 + s * 16 + l16) * BKP;
            bf.h[0] = *(const us8*)(br + 8 * half);
            bf.h[1] = *(const us8*)(br + 16 + 8 * half);
            #pragma unroll
            for (int m = 0; m < 2; ++m)
                acc[m][s] = __builtin_amdgcn_wmma_f32_16x16x32_bf16(
                    false, a[m].v, false, bf.v, (short)0, acc[m][s], false, false);
        }
        async_wait();
        __syncthreads();
        p ^= 1;
    }
}

// ---------------------------------------------------------------------------
// Kernel 1: E[b] = Xbf[b] * Xbf[b]^T  (M=N=512, K=4096). grid (4,4,16).
// ---------------------------------------------------------------------------
__global__ __launch_bounds__(256)
void cam_energy_kernel(const unsigned short* __restrict__ xbf,
                       float* __restrict__ energy) {
    const int b   = blockIdx.z;
    const int it0 = blockIdx.y * BMT;
    const int jt0 = blockIdx.x * BNT;
    const unsigned short* Xb = xbf + (size_t)b * CC * CN;
    float* Eb = energy + (size_t)b * CC * CC;

    __shared__ unsigned short smem[2 * BUF];
    v8f acc[2][4] = {};
    mm_core<CN>(Xb + (size_t)it0 * CN, CN, Xb + (size_t)jt0 * CN, CN, smem, acc);

    const int lane = threadIdx.x & 31, w = threadIdx.x >> 5;
    const int wm = w & 3, wn = w >> 2, half = lane >> 4, l16 = lane & 15;
    #pragma unroll
    for (int m = 0; m < 2; ++m)
        #pragma unroll
        for (int s = 0; s < 4; ++s)
            #pragma unroll
            for (int v = 0; v < 8; ++v) {
                const int row = it0 + wm * 32 + m * 16 + v + 8 * half;
                const int col = jt0 + wn * 64 + s * 16 + l16;
                Eb[(size_t)row * CC + col] = acc[m][s][v];
            }
}

// ---------------------------------------------------------------------------
// Kernel 2: attention = softmax(rowmax - E) == softmax(-E); write bf16.
// One block (256 thr) per 512-row; a_j = exp(minE - E_j) / sum.
// ---------------------------------------------------------------------------
__global__ __launch_bounds__(256)
void cam_softmax_kernel(const float* __restrict__ e,
                        unsigned short* __restrict__ attn) {
    const float* row = e + (size_t)blockIdx.x * CC;
    unsigned short* arow = attn + (size_t)blockIdx.x * CC;
    const int t = threadIdx.x;
    __shared__ float red[256];

    const float e0 = row[t], e1 = row[t + 256];
    red[t] = fminf(e0, e1);
    __syncthreads();
    #pragma unroll
    for (int s = 128; s > 0; s >>= 1) {
        if (t < s) red[t] = fminf(red[t], red[t + s]);
        __syncthreads();
    }
    const float mn = red[0];
    __syncthreads();

    const float w0 = __expf(mn - e0);
    const float w1 = __expf(mn - e1);
    red[t] = w0 + w1;
    __syncthreads();
    #pragma unroll
    for (int s = 128; s > 0; s >>= 1) {
        if (t < s) red[t] += red[t + s];
        __syncthreads();
    }
    const float inv = 1.0f / red[0];
    arow[t]       = f2bf(w0 * inv);
    arow[t + 256] = f2bf(w1 * inv);
}

// ---------------------------------------------------------------------------
// Kernel 3: out = gamma * (Attn @ X) + x  (M=512, N=4096, K=512).
// B operand comes from Xt (bf16 transposed): rows = spatial n, K-contiguous.
// grid (32, 4, 16).
// ---------------------------------------------------------------------------
__global__ __launch_bounds__(256)
void cam_out_kernel(const float* __restrict__ x,
                    const unsigned short* __restrict__ attn,
                    const unsigned short* __restrict__ xt,
                    const float* __restrict__ gamma,
                    float* __restrict__ out) {
    const int b   = blockIdx.z;
    const int it0 = blockIdx.y * BMT;   // channel rows
    const int nt0 = blockIdx.x * BNT;   // spatial cols
    const float* xb = x + (size_t)b * CC * CN;
    const unsigned short* Ab  = attn + (size_t)b * CC * CC;
    const unsigned short* Xtb = xt   + (size_t)b * CN * CC;
    float* ob = out + (size_t)b * CC * CN;

    __shared__ unsigned short smem[2 * BUF];
    v8f acc[2][4] = {};
    mm_core<CC>(Ab + (size_t)it0 * CC, CC, Xtb + (size_t)nt0 * CC, CC, smem, acc);

    const float g = gamma[0];
    const int lane = threadIdx.x & 31, w = threadIdx.x >> 5;
    const int wm = w & 3, wn = w >> 2, half = lane >> 4, l16 = lane & 15;
    #pragma unroll
    for (int m = 0; m < 2; ++m)
        #pragma unroll
        for (int s = 0; s < 4; ++s)
            #pragma unroll
            for (int v = 0; v < 8; ++v) {
                const int row = it0 + wm * 32 + m * 16 + v + 8 * half;
                const int col = nt0 + wn * 64 + s * 16 + l16;
                const size_t o = (size_t)row * CN + col;
                ob[o] = g * acc[m][s][v] + xb[o];
            }
}

// ---------------------------------------------------------------------------
extern "C" void kernel_launch(void* const* d_in, const int* in_sizes, int n_in,
                              void* d_out, int out_size, void* d_ws, size_t ws_size,
                              hipStream_t stream) {
    (void)in_sizes; (void)n_in; (void)out_size; (void)ws_size;
    const float* x     = (const float*)d_in[0];
    const float* gamma = (const float*)d_in[1];
    float*       out   = (float*)d_out;

    // Workspace layout (152 MB total):
    //   [0,   64M) bf16 X   (B x C x N)
    //   [64M, 128M) bf16 X^T (B x N x C)
    //   [128M,144M) fp32 energy (B x C x C)
    //   [144M,152M) bf16 attention (B x C x C)
    char* ws = (char*)d_ws;
    unsigned short* xbf    = (unsigned short*)ws;
    unsigned short* xt     = (unsigned short*)(ws + (size_t)64 * 1024 * 1024);
    float*          energy = (float*)         (ws + (size_t)128 * 1024 * 1024);
    unsigned short* attn   = (unsigned short*)(ws + (size_t)144 * 1024 * 1024);

    dim3 g0(CN / 64, CC / 64, CB);          // (64, 8, 16)
    cam_convert_kernel<<<g0, 256, 0, stream>>>(x, xbf, xt);

    dim3 g1(CC / BNT, CC / BMT, CB);        // (4, 4, 16)
    cam_energy_kernel<<<g1, 256, 0, stream>>>(xbf, energy);

    cam_softmax_kernel<<<CB * CC, 256, 0, stream>>>(energy, attn);

    dim3 g3(CN / BNT, CC / BMT, CB);        // (32, 4, 16)
    cam_out_kernel<<<g3, 256, 0, stream>>>(x, attn, xt, gamma, out);
}
